// MaskLayer_16166256902681
// MI455X (gfx1250) — matrix operand order repaired and checked by reading, not compiled, not run on hardware
//
#include <hip/hip_runtime.h>
#include <hip/hip_bf16.h>

// ---------------------------------------------------------------------------
// Fused ROI mask head for MI455X (gfx1250, wave32, WMMA).
// One workgroup (256 thr = 8 waves) per ROI; everything lives in LDS:
//   sA: padded crop  [34][34][64] bf16  (147,968 B)  -> reused as conv2 out [34][34][48]
//   sB: activations  [34][34][48] bf16  (110,976 B)
// Total ~253 KB LDS -- only possible on CDNA5 (320 KB/WGP).
// Convs = implicit GEMM via v_wmma_f32_16x16x32_bf16 (f32 accumulate).
// ---------------------------------------------------------------------------

typedef __attribute__((ext_vector_type(16))) __bf16 v16bf;
typedef __attribute__((ext_vector_type(8)))  __bf16 v8bf;
typedef __attribute__((ext_vector_type(8)))  float  v8f;

union Frag { v16bf v; v8bf h[2]; };

__device__ __forceinline__ unsigned short f2bf(float f) {
    unsigned u = __float_as_uint(f);
    u += 0x7FFFu + ((u >> 16) & 1u);          // round-to-nearest-even
    return (unsigned short)(u >> 16);
}
__device__ __forceinline__ float bf2f(unsigned short h) {
    return __uint_as_float(((unsigned)h) << 16);
}

// 3x3 conv, Cin = CS_IN (64 or 48), Cout = 48, SAME padding, +bias +ReLU.
// sIn:  padded input  [34][34][CS_IN] bf16 (halo zero)
// sOut: padded output [34][34][48]    bf16 (interior ch0..47 written)
// wp:   prepped weights [9][48][64] bf16 (k zero-padded to 64)
// HALF_K1: second K-chunk has only 16 valid channels (Cin==48) -> zero A half.
template<int CS_IN, bool HALF_K1>
__device__ __forceinline__ void conv3x3_wmma(
    const unsigned short* sIn, unsigned short* sOut,
    const unsigned short* __restrict__ wp, const float* __restrict__ bias,
    int lane, int wave)
{
    const int hi = lane >> 4;          // high half-wave
    const int ln = lane & 15;
    v8bf zero8;
#pragma unroll
    for (int i = 0; i < 8; ++i) zero8[i] = (__bf16)0.0f;
    const v8f zf = {0.f, 0.f, 0.f, 0.f, 0.f, 0.f, 0.f, 0.f};

#pragma unroll 1
    for (int n = 0; n < 3; ++n) {                       // 3 N-tiles of 16 out-ch
        v8f acc[8];
#pragma unroll
        for (int m = 0; m < 8; ++m) acc[m] = zf;

#pragma unroll 1
        for (int tap = 0; tap < 9; ++tap) {             // 3x3 taps
            const int dy = tap / 3, dx = tap % 3;
#pragma unroll
            for (int kc = 0; kc < 2; ++kc) {            // K chunks of 32
                const int kb = kc * 32;
                // B fragment: lane = column (out-ch), K = 8*hi + s, 16 + 8*hi + s
                Frag bfr;
                const unsigned short* wb =
                    wp + (((tap * 48) + n * 16 + ln) * 64 + kb + hi * 8);
                bfr.h[0] = *(const v8bf*)(wb);
                bfr.h[1] = *(const v8bf*)(wb + 16);
                // 8 M-tiles (16 pixels each) share this B fragment
#pragma unroll
                for (int m = 0; m < 8; ++m) {
                    const int t  = wave * 8 + m;        // tile id 0..63
                    const int y  = t >> 1;
                    const int x0 = (t & 1) << 4;
                    const unsigned short* ap =
                        sIn + ((y + dy) * 34 + x0 + dx + ln) * CS_IN + kb + hi * 8;
                    Frag afr;
                    afr.h[0] = *(const v8bf*)(ap);
                    if (HALF_K1 && kc == 1) afr.h[1] = zero8;   // k 48..63 = 0
                    else                    afr.h[1] = *(const v8bf*)(ap + 16);
                    acc[m] = __builtin_amdgcn_wmma_f32_16x16x32_bf16(
                        false, afr.v, false, bfr.v, (short)0, acc[m], false, false);
                }
            }
        }
        // epilogue: bias + ReLU + bf16 store. C/D layout: M = v + 8*hi, N = ln.
        const float bch = bias[n * 16 + ln];
#pragma unroll
        for (int m = 0; m < 8; ++m) {
            const int t  = wave * 8 + m;
            const int y  = t >> 1;
            const int x0 = (t & 1) << 4;
#pragma unroll
            for (int v = 0; v < 8; ++v) {
                float r = acc[m][v] + bch;
                r = fmaxf(r, 0.0f);
                const int px = x0 + v + hi * 8;
                sOut[((y + 1) * 34 + (px + 1)) * 48 + n * 16 + ln] = f2bf(r);
            }
        }
    }
}

// Pre-swizzle conv weights (HWIO f32) into [tap][ncol=48][k=64] bf16, k padded.
__global__ void prep_weights(const float* __restrict__ w1,
                             const float* __restrict__ w2,
                             const float* __restrict__ w3,
                             unsigned short* __restrict__ wpack)
{
    const int e = blockIdx.x * 256 + threadIdx.x;
    if (e >= 3 * 27648) return;
    const int conv = e / 27648;
    const int rem  = e % 27648;
    const int tap  = rem / (48 * 64);
    const int rem2 = rem % (48 * 64);
    const int nc   = rem2 / 64;
    const int k    = rem2 % 64;
    float v = 0.f;
    if (conv == 0) {
        v = w1[(tap * 64 + k) * 48 + nc];                 // Cin = 64
    } else if (k < 48) {
        const float* w = (conv == 1) ? w2 : w3;           // Cin = 48, pad k>=48
        v = w[(tap * 48 + k) * 48 + nc];
    }
    wpack[e] = f2bf(v);
}

__global__ __launch_bounds__(256, 1) void mask_head_fused(
    const float* __restrict__ features, const float* __restrict__ bboxes,
    const int* __restrict__ cls, const unsigned short* __restrict__ wpack,
    const float* __restrict__ b1, const float* __restrict__ b2,
    const float* __restrict__ b3, const float* __restrict__ w_out,
    const float* __restrict__ b_out, float* __restrict__ out)
{
    __shared__ __align__(16) unsigned short sA[34 * 34 * 64];
    __shared__ __align__(16) unsigned short sB[34 * 34 * 48];
    __shared__ float s_wsel[48];
    __shared__ float s_bout;

    const int tid  = threadIdx.x;
    const int lane = tid & 31;
    const int wave = tid >> 5;
    const int roi  = blockIdx.x;           // 0..1023
    const int b    = roi >> 7;             // batch (N=128)

    // ---- phase 0: zero LDS (halo + pad channels), stage class weights ----
    {
        uint4 z; z.x = z.y = z.z = z.w = 0u;
        uint4* pa = (uint4*)sA;
        for (int i = tid; i < 34 * 34 * 64 / 8; i += 256) pa[i] = z;
        uint4* pb = (uint4*)sB;
        for (int i = tid; i < 34 * 34 * 48 / 8; i += 256) pb[i] = z;
    }
    const int cl = cls[b];
    if (tid < 48) s_wsel[tid] = w_out[tid * 3 + cl];
    if (tid == 48) s_bout = b_out[cl];
    __syncthreads();

    // ---- phase 1: bilinear crop_and_resize into sA interior (bf16) ----
    const float* bx = bboxes + roi * 4;
    const float ya = rintf(bx[0] * 256.f);
    const float xa = rintf(bx[1] * 256.f);
    const float yb = rintf(bx[2] * 256.f) - 1.f;
    const float xb = rintf(bx[3] * 256.f) - 1.f;
#pragma unroll
    for (int i = 0; i < 4; ++i) {
        const int p  = tid + i * 256;      // 1024 pixels / 256 threads
        const int py = p >> 5, px = p & 31;
        const float ys = ya + (yb - ya) * (py * (1.f / 31.f));
        const float xs = xa + (xb - xa) * (px * (1.f / 31.f));
        int y0 = (int)floorf(ys); y0 = y0 < 0 ? 0 : (y0 > 255 ? 255 : y0);
        int x0 = (int)floorf(xs); x0 = x0 < 0 ? 0 : (x0 > 255 ? 255 : x0);
        const int y1 = (y0 + 1 > 255) ? 255 : y0 + 1;
        const int x1 = (x0 + 1 > 255) ? 255 : x0 + 1;
        const float wy  = ys - (float)y0;
        const float wx  = xs - (float)x0;
        const float w00 = (1.f - wy) * (1.f - wx), w01 = (1.f - wy) * wx;
        const float w10 = wy * (1.f - wx),         w11 = wy * wx;
        const size_t base = (size_t)b * 256 * 256 * 64;
        const float4* f00 = (const float4*)(features + base + ((size_t)y0 * 256 + x0) * 64);
        const float4* f01 = (const float4*)(features + base + ((size_t)y0 * 256 + x1) * 64);
        const float4* f10 = (const float4*)(features + base + ((size_t)y1 * 256 + x0) * 64);
        const float4* f11 = (const float4*)(features + base + ((size_t)y1 * 256 + x1) * 64);
        unsigned short* dst = sA + ((py + 1) * 34 + px + 1) * 64;
#pragma unroll 4
        for (int cc = 0; cc < 16; ++cc) {
            const float4 a = f00[cc], bb = f01[cc], c = f10[cc], d = f11[cc];
            const unsigned short r0 = f2bf(a.x * w00 + bb.x * w01 + c.x * w10 + d.x * w11);
            const unsigned short r1 = f2bf(a.y * w00 + bb.y * w01 + c.y * w10 + d.y * w11);
            const unsigned short r2 = f2bf(a.z * w00 + bb.z * w01 + c.z * w10 + d.z * w11);
            const unsigned short r3 = f2bf(a.w * w00 + bb.w * w01 + c.w * w10 + d.w * w11);
            uint2 pk;
            pk.x = (unsigned)r0 | ((unsigned)r1 << 16);
            pk.y = (unsigned)r2 | ((unsigned)r3 << 16);
            *(uint2*)(dst + cc * 4) = pk;
        }
    }
    __syncthreads();

    // ---- phase 2: conv1 64->48 (sA -> sB) ----
    conv3x3_wmma<64, false>(sA, sB, wpack, b1, lane, wave);
    __syncthreads();

    // ---- phase 3: re-zero sA halo in [34][34][48] layout, conv2 (sB -> sA) ----
    for (int i = tid; i < 132; i += 256) {
        int py, px;
        if      (i < 34)  { py = 0;            px = i;        }
        else if (i < 68)  { py = 33;           px = i - 34;   }
        else if (i < 100) { py = i - 68 + 1;   px = 0;        }
        else              { py = i - 100 + 1;  px = 33;       }
        uint4 z; z.x = z.y = z.z = z.w = 0u;
        uint4* q = (uint4*)(sA + (py * 34 + px) * 48);
#pragma unroll
        for (int j = 0; j < 6; ++j) q[j] = z;
    }
    conv3x3_wmma<48, true>(sB, sA, wpack + 27648, b2, lane, wave);
    __syncthreads();

    // ---- phase 4: conv3 48->48 (sA -> sB) ----
    conv3x3_wmma<48, true>(sA, sB, wpack + 2 * 27648, b3, lane, wave);
    __syncthreads();

    // ---- phase 5: 1x1 conv with class selection -> f32 output ----
    const size_t obase = (size_t)roi * 1024;
#pragma unroll
    for (int i = 0; i < 4; ++i) {
        const int p  = tid + i * 256;
        const int py = p >> 5, px = p & 31;
        const unsigned short* src = sB + ((py + 1) * 34 + px + 1) * 48;
        float acc = s_bout;
#pragma unroll
        for (int c = 0; c < 48; ++c) acc += bf2f(src[c]) * s_wsel[c];
        out[obase + p] = acc;
    }
}

extern "C" void kernel_launch(void* const* d_in, const int* in_sizes, int n_in,
                              void* d_out, int out_size, void* d_ws, size_t ws_size,
                              hipStream_t stream)
{
    (void)in_sizes; (void)n_in; (void)out_size; (void)ws_size;
    const float* features = (const float*)d_in[0];
    const float* bboxes   = (const float*)d_in[1];
    /* d_in[2] = scores: unused by reference output */
    const int*   cls      = (const int*)d_in[3];
    const float* w1       = (const float*)d_in[4];
    const float* b1       = (const float*)d_in[5];
    const float* w2       = (const float*)d_in[6];
    const float* b2       = (const float*)d_in[7];
    const float* w3       = (const float*)d_in[8];
    const float* b3       = (const float*)d_in[9];
    const float* w_out    = (const float*)d_in[10];
    const float* b_out    = (const float*)d_in[11];

    unsigned short* wpack = (unsigned short*)d_ws;   // 3*9*48*64 bf16 = 165,888 B

    prep_weights<<<(3 * 27648 + 255) / 256, 256, 0, stream>>>(w1, w2, w3, wpack);
    mask_head_fused<<<1024, 256, 0, stream>>>(features, bboxes, cls, wpack,
                                              b1, b2, b3, w_out, b_out,
                                              (float*)d_out);
}
